// GraphTransformer_23579370455378
// MI455X (gfx1250) — compile-verified, matrix operand order checked by reading, not run on hardware
//
#include <hip/hip_runtime.h>
#include <hip/hip_bf16.h>
#include <math.h>

typedef __attribute__((ext_vector_type(16))) _Float16 v16h;
typedef __attribute__((ext_vector_type(8)))  float    v8f;

#define NNODES  20000
#define NEDGES  640000
#define DD      256
#define NGROUPS 5
#define TOKD    1024
#define REDB    512

// ---------------------------------------------------------------------------
// Fragment packers: convert row-major f32 matrices into the exact VGPR
// fragment layout of V_WMMA_F32_16X16X32_F16 (CDNA5 ISA 7.12.2).
// A (16x32, f16): lane 0-15 -> M=lane, lanes 16-31 -> M=lane-16 with +8 K skew;
//   half-element e -> K = e + (e>=8 ? 8 : 0) + (lane>=16 ? 8 : 0)
// B (32x16, f16): symmetric, N = lane % 16, same K mapping.
// Fragment storage: [tile][lane][e] contiguous -> wave loads v16h per lane.
// ---------------------------------------------------------------------------
__global__ void pack_a_frag(const float* __restrict__ src, _Float16* __restrict__ dst,
                            int M, int Kd) {
    int KT = Kd >> 5;
    size_t total = (size_t)M * Kd;
    for (size_t idx = (size_t)blockIdx.x * blockDim.x + threadIdx.x; idx < total;
         idx += (size_t)gridDim.x * blockDim.x) {
        int e    = (int)(idx & 15);
        int lane = (int)((idx >> 4) & 31);
        size_t tile = idx >> 9;
        int kt = (int)(tile % KT);
        int mt = (int)(tile / KT);
        int dm = lane & 15;
        int dk = e + ((e & 8) ? 8 : 0) + ((lane & 16) ? 8 : 0);
        dst[idx] = (_Float16)src[(size_t)(mt * 16 + dm) * Kd + kt * 32 + dk];
    }
}

__global__ void pack_b_frag(const float* __restrict__ src, _Float16* __restrict__ dst,
                            int Kd, int Nd) {
    int KT = Kd >> 5;
    size_t total = (size_t)Kd * Nd;
    for (size_t idx = (size_t)blockIdx.x * blockDim.x + threadIdx.x; idx < total;
         idx += (size_t)gridDim.x * blockDim.x) {
        int e    = (int)(idx & 15);
        int lane = (int)((idx >> 4) & 31);
        size_t tile = idx >> 9;
        int kt = (int)(tile % KT);
        int nt = (int)(tile / KT);
        int dn = lane & 15;
        int dk = e + ((e & 8) ? 8 : 0) + ((lane & 16) ? 8 : 0);
        dst[idx] = (_Float16)src[(size_t)(kt * 32 + dk) * Nd + nt * 16 + dn];
    }
}

// ---------------------------------------------------------------------------
// WMMA GEMM: out[M,N] = Ah[M,K] * Bh[K,N] + bias (+ res).
// One wave computes a 16x64 strip: the A fragment is loaded once per K-step
// and reused across 4 WMMAs (4 accumulators) -> 4 v_wmma per 5 fragment
// loads, 4x less A traffic than one-tile-per-wave.
// ---------------------------------------------------------------------------
__global__ void wmma_gemm(const _Float16* __restrict__ Ah, const _Float16* __restrict__ Bh,
                          const float* __restrict__ bias, const float* __restrict__ res,
                          float* __restrict__ out, int MT, int NG, int KT, int Nd) {
    int wave = (int)((blockIdx.x * blockDim.x + threadIdx.x) >> 5);
    if (wave >= MT * NG) return;               // wave-uniform: EXEC all-1 for WMMA
    int lane = threadIdx.x & 31;
    int mt = wave / NG, ng = wave % NG;
    const v16h* Ap = (const v16h*)(Ah + (size_t)mt * KT * 512 + lane * 16);
    const v16h* Bp = (const v16h*)(Bh + (size_t)(ng * 4) * KT * 512 + lane * 16);
    v8f acc0 = {}, acc1 = {}, acc2 = {}, acc3 = {};
    for (int kt = 0; kt < KT; ++kt) {
        v16h a = Ap[kt * 32];
        if (kt + 1 < KT) __builtin_prefetch((const void*)(Ap + (kt + 1) * 32), 0, 3);
        v16h b0 = Bp[(0 * KT + kt) * 32];
        v16h b1 = Bp[(1 * KT + kt) * 32];
        v16h b2 = Bp[(2 * KT + kt) * 32];
        v16h b3 = Bp[(3 * KT + kt) * 32];
        acc0 = __builtin_amdgcn_wmma_f32_16x16x32_f16(false, a, false, b0, (short)0, acc0, false, false);
        acc1 = __builtin_amdgcn_wmma_f32_16x16x32_f16(false, a, false, b1, (short)0, acc1, false, false);
        acc2 = __builtin_amdgcn_wmma_f32_16x16x32_f16(false, a, false, b2, (short)0, acc2, false, false);
        acc3 = __builtin_amdgcn_wmma_f32_16x16x32_f16(false, a, false, b3, (short)0, acc3, false, false);
    }
    int rowb = mt * 16 + ((lane & 16) ? 8 : 0);
    int cb = lane & 15;
    v8f accs[4] = {acc0, acc1, acc2, acc3};
#pragma unroll
    for (int j = 0; j < 4; ++j) {
        int col = (ng * 4 + j) * 16 + cb;
        float bvv = bias ? bias[col] : 0.0f;
#pragma unroll
        for (int r = 0; r < 8; ++r) {
            size_t o = (size_t)(rowb + r) * Nd + col;
            float v = accs[j][r] + bvv;
            if (res) v += res[o];
            out[o] = v;
        }
    }
}

// ---------------------------------------------------------------------------
// Irregular graph pieces
// ---------------------------------------------------------------------------
__global__ void gather_rows(const int* __restrict__ ids, const float* __restrict__ emb,
                            float* __restrict__ x) {
    int i = blockIdx.x * blockDim.x + threadIdx.x;        // one float4 each
    if (i >= NNODES * 64) return;
    int node = i >> 6, j = i & 63;
    ((float4*)x)[i] = ((const float4*)(emb + (size_t)ids[node] * DD))[j];
}

__global__ void zero_f4(float4* __restrict__ p, int n4) {
    int i = blockIdx.x * blockDim.x + threadIdx.x;
    if (i < n4) p[i] = make_float4(0.f, 0.f, 0.f, 0.f);
}

__global__ void sent_proj(const float* __restrict__ sent, const float* __restrict__ Ws,
                          const float* __restrict__ bs, float* __restrict__ S) {
    int g = blockIdx.x, d = threadIdx.x;
    float acc = bs[d];
    for (int k = 0; k < TOKD; ++k) acc += sent[g * TOKD + k] * Ws[(size_t)k * DD + d];
    S[g * DD + d] = acc;
}

__global__ void edge_alpha(const int* __restrict__ ei, const int* __restrict__ batch,
                           const float* __restrict__ Q, const float* __restrict__ K,
                           const float* __restrict__ S, float* __restrict__ alpha) {
    int e = blockIdx.x * blockDim.x + threadIdx.x;
    if (e >= NEDGES) return;
    int r = ei[e], c = ei[NEDGES + e];
    int gr = batch[r] / 5, gc = batch[c] / 5;
    const float4* q  = (const float4*)(Q + (size_t)r * DD);
    const float4* k  = (const float4*)(K + (size_t)c * DD);
    const float4* sr = (const float4*)(S + gr * DD);
    const float4* sc = (const float4*)(S + gc * DD);
    for (int h = 0; h < 4; ++h) {
        float qk = 0.f, sk = 0.f, qs = 0.f;
#pragma unroll
        for (int i = 0; i < 16; ++i) {
            float4 qv = q[h * 16 + i], kv = k[h * 16 + i];
            float4 sv = sr[h * 16 + i], cv = sc[h * 16 + i];
            qk += qv.x * kv.x + qv.y * kv.y + qv.z * kv.z + qv.w * kv.w;
            sk += sv.x * kv.x + sv.y * kv.y + sv.z * kv.z + sv.w * kv.w;
            qs += qv.x * cv.x + qv.y * cv.y + qv.z * cv.z + qv.w * cv.w;
        }
        float a = (qk + sk + qs) * (1.0f / 24.0f);          // 3*sqrt(64)
        alpha[(size_t)h * NEDGES + e] = (a > 0.f) ? a : 0.2f * a;  // leaky_relu
    }
}

__global__ void red_max(const float* __restrict__ alpha, float* __restrict__ pmax) {
    __shared__ float sm[256];
    float m[4] = {-1e30f, -1e30f, -1e30f, -1e30f};
    for (int e = blockIdx.x * blockDim.x + threadIdx.x; e < NEDGES;
         e += gridDim.x * blockDim.x)
#pragma unroll
        for (int h = 0; h < 4; ++h) m[h] = fmaxf(m[h], alpha[(size_t)h * NEDGES + e]);
    for (int h = 0; h < 4; ++h) {
        sm[threadIdx.x] = m[h]; __syncthreads();
        for (int o = 128; o > 0; o >>= 1) {
            if (threadIdx.x < o) sm[threadIdx.x] = fmaxf(sm[threadIdx.x], sm[threadIdx.x + o]);
            __syncthreads();
        }
        if (threadIdx.x == 0) pmax[h * gridDim.x + blockIdx.x] = sm[0];
        __syncthreads();
    }
}

__global__ void red_sum(const float* __restrict__ alpha, const float* __restrict__ gms,
                        float* __restrict__ psum) {
    __shared__ float sm[256];
    float s[4] = {0.f, 0.f, 0.f, 0.f};
    for (int e = blockIdx.x * blockDim.x + threadIdx.x; e < NEDGES;
         e += gridDim.x * blockDim.x)
#pragma unroll
        for (int h = 0; h < 4; ++h) s[h] += __expf(alpha[(size_t)h * NEDGES + e] - gms[h]);
    for (int h = 0; h < 4; ++h) {
        sm[threadIdx.x] = s[h]; __syncthreads();
        for (int o = 128; o > 0; o >>= 1) {
            if (threadIdx.x < o) sm[threadIdx.x] += sm[threadIdx.x + o];
            __syncthreads();
        }
        if (threadIdx.x == 0) psum[h * gridDim.x + blockIdx.x] = sm[0];
        __syncthreads();
    }
}

__global__ void red_final(const float* __restrict__ p, float* __restrict__ g,
                          int nb, int isSum) {
    __shared__ float sm[256];
    for (int h = 0; h < 4; ++h) {
        float acc = isSum ? 0.f : -1e30f;
        for (int i = threadIdx.x; i < nb; i += 256)
            acc = isSum ? (acc + p[h * nb + i]) : fmaxf(acc, p[h * nb + i]);
        sm[threadIdx.x] = acc; __syncthreads();
        for (int o = 128; o > 0; o >>= 1) {
            if (threadIdx.x < o)
                sm[threadIdx.x] = isSum ? (sm[threadIdx.x] + sm[threadIdx.x + o])
                                        : fmaxf(sm[threadIdx.x], sm[threadIdx.x + o]);
            __syncthreads();
        }
        if (threadIdx.x == 0) g[(isSum ? 4 : 0) + h] = sm[0];
        __syncthreads();
    }
}

__global__ void edge_scatter(const int* __restrict__ ei, const float* __restrict__ alpha,
                             const float* __restrict__ gms, const float* __restrict__ V,
                             float* __restrict__ agg) {
    int e = blockIdx.x * blockDim.x + threadIdx.x;
    if (e >= NEDGES) return;
    int r = ei[e], c = ei[NEDGES + e];
    float w[4];
#pragma unroll
    for (int h = 0; h < 4; ++h)
        w[h] = __expf(alpha[(size_t)h * NEDGES + e] - gms[h]) / gms[4 + h];
    const float4* v = (const float4*)(V + (size_t)c * DD);
    float* a = agg + (size_t)r * DD;
    for (int i = 0; i < 64; ++i) {
        float4 vv = v[i];
        float ww = w[i >> 4];
        atomicAdd(a + i * 4 + 0, vv.x * ww);
        atomicAdd(a + i * 4 + 1, vv.y * ww);
        atomicAdd(a + i * 4 + 2, vv.z * ww);
        atomicAdd(a + i * 4 + 3, vv.w * ww);
    }
}

// ---------------------------------------------------------------------------
// Collapsed tail (mathematically identical to the reference):
// sel2[b,i,:] == A[b,0,:] for all i, so the output is one 1024-vector
// broadcast to all 25 graphs.
// ---------------------------------------------------------------------------
__global__ void x1_row0(const float* __restrict__ x, const float* __restrict__ W1,
                        const float* __restrict__ b1, float* __restrict__ x1) {
    int d = threadIdx.x;
    float acc = b1[d];
    for (int k = 0; k < DD; ++k) acc += x[k] * W1[(size_t)k * DD + d];
    x1[d] = acc;
}

__global__ void attn_pool(const float* __restrict__ x1, const float* __restrict__ tok,
                          float* __restrict__ A) {
    int b = blockIdx.x, t = threadIdx.x;
    __shared__ float w[256];
    __shared__ float red[256];
    const float* tb = tok + (size_t)b * 256 * DD;
    float s = 0.f;
    for (int d = 0; d < DD; ++d) s += x1[d] * tb[(size_t)t * DD + d];
    s *= (1.0f / 16.0f);                       // / sqrt(256)
    red[t] = s; __syncthreads();
    for (int o = 128; o > 0; o >>= 1) {
        if (t < o) red[t] = fmaxf(red[t], red[t + o]);
        __syncthreads();
    }
    float mx = red[0]; __syncthreads();
    float ex = __expf(s - mx);
    w[t] = ex; red[t] = ex; __syncthreads();
    for (int o = 128; o > 0; o >>= 1) {
        if (t < o) red[t] += red[t + o];
        __syncthreads();
    }
    float inv = 1.0f / red[0];
    __syncthreads();
    float acc = 0.f;                           // second phase: t plays role of d
    for (int tt = 0; tt < 256; ++tt) acc += w[tt] * tb[(size_t)tt * DD + t];
    A[b * DD + t] = acc * inv;
}

__global__ void out_proj(const float* __restrict__ A, const float* __restrict__ Wout,
                         const float* __restrict__ bout, float* __restrict__ xr) {
    int d = threadIdx.x;
    float acc = bout[d];
    for (int k = 0; k < 5 * DD; ++k) acc += A[k] * Wout[(size_t)k * DD + d];
    xr[d] = acc;
}

__global__ void final_proj(const float* __restrict__ xr, const float* __restrict__ W3,
                           const float* __restrict__ b3, float* __restrict__ out) {
    int v = blockIdx.x * blockDim.x + threadIdx.x;
    if (v >= TOKD) return;
    float acc = b3[v];
    for (int d = 0; d < DD; ++d) acc += xr[d] * W3[(size_t)d * TOKD + v];
    for (int g = 0; g < 25; ++g) out[(size_t)g * TOKD + v] = acc;
}

// ---------------------------------------------------------------------------
extern "C" void kernel_launch(void* const* d_in, const int* in_sizes, int n_in,
                              void* d_out, int out_size, void* d_ws, size_t ws_size,
                              hipStream_t stream) {
    const int*   concept_ids = (const int*)d_in[0];
    const int*   edge_index  = (const int*)d_in[1];
    const float* sent        = (const float*)d_in[3];
    const float* token_emb   = (const float*)d_in[4];
    const int*   batch       = (const int*)d_in[5];
    const float* ent_emb     = (const float*)d_in[6];
    const float* Wq = (const float*)d_in[7];  const float* bq = (const float*)d_in[8];
    const float* Wk = (const float*)d_in[9];  const float* bk = (const float*)d_in[10];
    const float* Wv = (const float*)d_in[11]; const float* bv = (const float*)d_in[12];
    const float* Ws = (const float*)d_in[13]; const float* bs = (const float*)d_in[14];
    const float* Wo = (const float*)d_in[15]; const float* bo = (const float*)d_in[16];
    const float* W1 = (const float*)d_in[17]; const float* b1 = (const float*)d_in[18];
    const float* W2 = (const float*)d_in[19]; const float* b2 = (const float*)d_in[20];
    const float* W3 = (const float*)d_in[21]; const float* b3 = (const float*)d_in[22];
    const float* Wout = (const float*)d_in[23]; const float* bout = (const float*)d_in[24];
    float* out = (float*)d_out;

    // ---- workspace layout -------------------------------------------------
    char* ws = (char*)d_ws;
    size_t off = 0;
    auto alloc = [&](size_t bytes) -> void* {
        off = (off + 255) & ~(size_t)255;
        void* p = ws + off;
        off += bytes;
        return p;
    };
    float*    xf32  = (float*)   alloc((size_t)NNODES * DD * 4);
    _Float16* xh    = (_Float16*)alloc((size_t)NNODES * DD * 2);
    float*    Qf    = (float*)   alloc((size_t)NNODES * DD * 4);   // reused as agg
    float*    Kf    = (float*)   alloc((size_t)NNODES * DD * 4);
    float*    Vf    = (float*)   alloc((size_t)NNODES * DD * 4);
    float*    alpha = (float*)   alloc((size_t)4 * NEDGES * 4);
    _Float16* wbh   = (_Float16*)alloc((size_t)TOKD * DD * 2);     // largest B frag
    _Float16* tokAh = (_Float16*)alloc((size_t)5 * 256 * TOKD * 2);
    float*    tokf  = (float*)   alloc((size_t)5 * 256 * DD * 4);
    float*    Sbuf  = (float*)   alloc((size_t)NGROUPS * DD * 4);
    float*    pbuf  = (float*)   alloc((size_t)4 * REDB * 4);
    float*    gms   = (float*)   alloc(8 * 4);
    float*    x1b   = (float*)   alloc(DD * 4);
    float*    Abuf  = (float*)   alloc((size_t)NGROUPS * DD * 4);
    float*    xrb   = (float*)   alloc(DD * 4);
    (void)ws_size; (void)in_sizes; (void)n_in; (void)out_size;

    const int MT = NNODES / 16, NG = DD / 64, KT = DD / 32;  // 1250, 4, 8
    const int gemmWaves = MT * NG;                            // 5000
    dim3 gemmGrid((gemmWaves + 3) / 4), gemmBlk(128);
    dim3 packGrid(4096), b256(256);
    dim3 edgeGrid((NEDGES + 255) / 256);
    dim3 f4Grid((NNODES * 64 + 255) / 256);

    // x = entity_embeddings[concept_ids]
    gather_rows<<<f4Grid, b256, 0, stream>>>(concept_ids, ent_emb, xf32);

    for (int l = 0; l < 3; ++l) {
        const float* Wql = Wq + (size_t)l * DD * DD;   const float* bql = bq + l * DD;
        const float* Wkl = Wk + (size_t)l * DD * DD;   const float* bkl = bk + l * DD;
        const float* Wvl = Wv + (size_t)l * DD * DD;   const float* bvl = bv + l * DD;
        const float* Wsl = Ws + (size_t)l * TOKD * DD; const float* bsl = bs + l * DD;
        const float* Wol = Wo + (size_t)l * DD * DD;   const float* bol = bo + l * DD;

        pack_a_frag<<<packGrid, b256, 0, stream>>>(xf32, xh, NNODES, DD);

        pack_b_frag<<<packGrid, b256, 0, stream>>>(Wql, wbh, DD, DD);
        wmma_gemm<<<gemmGrid, gemmBlk, 0, stream>>>(xh, wbh, bql, nullptr, Qf, MT, NG, KT, DD);
        pack_b_frag<<<packGrid, b256, 0, stream>>>(Wkl, wbh, DD, DD);
        wmma_gemm<<<gemmGrid, gemmBlk, 0, stream>>>(xh, wbh, bkl, nullptr, Kf, MT, NG, KT, DD);
        pack_b_frag<<<packGrid, b256, 0, stream>>>(Wvl, wbh, DD, DD);
        wmma_gemm<<<gemmGrid, gemmBlk, 0, stream>>>(xh, wbh, bvl, nullptr, Vf, MT, NG, KT, DD);

        sent_proj<<<NGROUPS, b256, 0, stream>>>(sent, Wsl, bsl, Sbuf);

        edge_alpha<<<edgeGrid, b256, 0, stream>>>(edge_index, batch, Qf, Kf, Sbuf, alpha);
        red_max<<<REDB, b256, 0, stream>>>(alpha, pbuf);
        red_final<<<1, b256, 0, stream>>>(pbuf, gms, REDB, 0);
        red_sum<<<REDB, b256, 0, stream>>>(alpha, gms, pbuf);
        red_final<<<1, b256, 0, stream>>>(pbuf, gms, REDB, 1);

        float* agg = Qf;                                    // Q is dead now
        zero_f4<<<f4Grid, b256, 0, stream>>>((float4*)agg, NNODES * 64);
        edge_scatter<<<edgeGrid, b256, 0, stream>>>(edge_index, alpha, gms, Vf, agg);

        pack_a_frag<<<packGrid, b256, 0, stream>>>(agg, xh, NNODES, DD);
        pack_b_frag<<<packGrid, b256, 0, stream>>>(Wol, wbh, DD, DD);
        // x = agg @ Wo + bo + x   (residual read+write same address per thread)
        wmma_gemm<<<gemmGrid, gemmBlk, 0, stream>>>(xh, wbh, bol, xf32, xf32, MT, NG, KT, DD);
    }

    // tail: x1 row 0 only (indexing collapse), tok = token_embedding @ W2 + b2
    x1_row0<<<1, b256, 0, stream>>>(xf32, W1, b1, x1b);

    const int tMT = (5 * 256) / 16, tKT = TOKD / 32;         // 80, 32
    pack_a_frag<<<packGrid, b256, 0, stream>>>(token_emb, tokAh, 5 * 256, TOKD);
    pack_b_frag<<<packGrid, b256, 0, stream>>>(W2, wbh, TOKD, DD);
    dim3 tGrid((tMT * NG + 3) / 4);
    wmma_gemm<<<tGrid, gemmBlk, 0, stream>>>(tokAh, wbh, b2, nullptr, tokf, tMT, NG, tKT, DD);

    attn_pool<<<NGROUPS, b256, 0, stream>>>(x1b, tokf, Abuf);
    out_proj<<<1, b256, 0, stream>>>(Abuf, Wout, bout, xrb);
    final_proj<<<4, b256, 0, stream>>>(xrb, W3, b3, out);
}
